// UNetAttention_48747878810137
// MI455X (gfx1250) — compile-verified, matrix operand order checked by reading, not run on hardware
//
#include <hip/hip_runtime.h>
#include <hip/hip_bf16.h>

typedef __attribute__((ext_vector_type(16))) _Float16 v16h;
typedef __attribute__((ext_vector_type(8)))  _Float16 v8h;
typedef __attribute__((ext_vector_type(8)))  float    v8f;

#define C_DIM 512
#define N_PIX 1024
#define BATCH 16
#define GROUPS 32

// async 16B copy global -> LDS (GV mode, tracked by ASYNCcnt)
#define ASYNC_CP_B128(ldsOff, gptr)                                             \
  asm volatile("global_load_async_to_lds_b128 %0, %1, off"                      \
               :: "v"((uint32_t)(ldsOff)), "v"((uint64_t)(uintptr_t)(gptr))     \
               : "memory")
#define WAIT_ASYNC0() asm volatile("s_wait_asynccnt 0x0" ::: "memory")

// ---------------- weight f32 -> f16 convert ----------------
__global__ __launch_bounds__(256) void wcvt_kernel(const float* __restrict__ wq,
                                                   const float* __restrict__ wk,
                                                   const float* __restrict__ wv,
                                                   const float* __restrict__ wp,
                                                   _Float16* __restrict__ out) {
  const int i = blockIdx.x * 256 + threadIdx.x;  // 0 .. 262143
  out[i]          = (_Float16)wq[i];
  out[262144 + i] = (_Float16)wk[i];
  out[524288 + i] = (_Float16)wv[i];
  out[786432 + i] = (_Float16)wp[i];
}

// ---------------- GroupNorm: x[B,C,N] -> h_t[B,N,C] (f16, transposed) ----------------
__global__ __launch_bounds__(256) void groupnorm_kernel(const float* __restrict__ x,
                                                        const float* __restrict__ gamma,
                                                        const float* __restrict__ beta,
                                                        _Float16* __restrict__ h_t) {
  const int b = blockIdx.y, g = blockIdx.x;
  const float* xg = x + ((size_t)b * C_DIM + (size_t)g * 16) * N_PIX;
  float s = 0.f, s2 = 0.f;
  for (int i = threadIdx.x; i < 16 * N_PIX; i += 256) {
    const float v = xg[i];
    s += v; s2 += v * v;
  }
  __shared__ float red[16];
  for (int off = 16; off > 0; off >>= 1) {
    s  += __shfl_down(s,  off, 32);
    s2 += __shfl_down(s2, off, 32);
  }
  const int wid = threadIdx.x >> 5, ln = threadIdx.x & 31;
  if (ln == 0) { red[wid] = s; red[8 + wid] = s2; }
  __syncthreads();
  if (threadIdx.x == 0) {
    float a = 0.f, c = 0.f;
    for (int i = 0; i < 8; ++i) { a += red[i]; c += red[8 + i]; }
    red[0] = a; red[8] = c;
  }
  __syncthreads();
  s = red[0]; s2 = red[8];
  const float mu   = s * (1.f / 16384.f);
  const float var  = s2 * (1.f / 16384.f) - mu * mu;
  const float rinv = rsqrtf(var + 1e-5f);
  for (int i = threadIdx.x; i < 16 * N_PIX; i += 256) {
    const int cl = i >> 10;                  // local channel 0..15
    const int n  = i & (N_PIX - 1);          // pixel
    const int c  = g * 16 + cl;
    const float h = (xg[i] - mu) * rinv * gamma[c] + beta[c];
    h_t[((size_t)b * N_PIX + n) * C_DIM + c] = (_Float16)h;
  }
}

// ---------------- WMMA NT-GEMM: D[M,Nout] = A[M,K] * B[Nout,K]^T ----------------
enum { EP_F16N = 0, EP_F16T = 1, EP_F32N = 2, EP_RESID = 3 };

template <int MODE>
__global__ __launch_bounds__(256) void gemm_nt_kernel(
    const _Float16* __restrict__ A,  long long sA, int lda,
    const _Float16* __restrict__ Bm, long long sB, int ldb,
    void* __restrict__ OutV,         long long sOut, int ldo,
    const float* __restrict__ bias,
    const float* __restrict__ res,   long long sRes,
    int K, float scale) {
  constexpr int PAD = 40;                       // 32 K halves + 8 pad (80B rows, 16B aligned)
  constexpr int TILE_HALVES = 128 * PAD;        // one buffer
  __shared__ _Float16 As[2 * TILE_HALVES];      // double buffered
  __shared__ _Float16 Bs[2 * TILE_HALVES];

  const int bx = blockIdx.x * 128;  // output col base
  const int by = blockIdx.y * 128;  // output row base
  const int bz = blockIdx.z;
  A  += (size_t)bz * sA;
  Bm += (size_t)bz * sB;

  const int tid  = threadIdx.x;
  const int lane = tid & 31;
  const int wid  = tid >> 5;
  const int wm   = (wid & 3) * 32;   // wave row base in tile (4 wave-rows)
  const int wn   = (wid >> 2) * 64;  // wave col base in tile (2 wave-cols)
  const int lr   = lane & 15;
  const int lh   = lane >> 4;

  // per-thread async-copy assignment: chunks tid and tid+256 of the 512
  // 16B chunks per operand tile (row = chunk>>2, k8 = (chunk&3)*8 halves)
  const int r0  = tid >> 2;
  const int k80 = (tid & 3) << 3;
  const _Float16* gA0 = A  + (size_t)(by + r0)      * lda + k80;
  const _Float16* gA1 = A  + (size_t)(by + r0 + 64) * lda + k80;
  const _Float16* gB0 = Bm + (size_t)(bx + r0)      * ldb + k80;
  const _Float16* gB1 = Bm + (size_t)(bx + r0 + 64) * ldb + k80;
  const uint32_t lA0 = (uint32_t)(uintptr_t)&As[r0 * PAD + k80];  // low 32b = LDS offset
  const uint32_t lA1 = lA0 + 64 * PAD * 2;
  const uint32_t lB0 = (uint32_t)(uintptr_t)&Bs[r0 * PAD + k80];
  const uint32_t lB1 = lB0 + 64 * PAD * 2;
  constexpr uint32_t BUF_BYTES = TILE_HALVES * 2;

  v8f zero = {};
  v8f acc[2][4];
#pragma unroll
  for (int i = 0; i < 2; ++i)
#pragma unroll
    for (int j = 0; j < 4; ++j) acc[i][j] = zero;

  // prologue: DMA tile 0 into buffer 0
  ASYNC_CP_B128(lA0, gA0);
  ASYNC_CP_B128(lA1, gA1);
  ASYNC_CP_B128(lB0, gB0);
  ASYNC_CP_B128(lB1, gB1);

  const int nTiles = K >> 5;  // K / 32
  for (int t = 0; t < nTiles; ++t) {
    const int cur = t & 1;
    WAIT_ASYNC0();        // my tile-t chunks are in LDS
    __syncthreads();      // everyone's tile-t in LDS; everyone done computing t-1

    if (t + 1 < nTiles) { // DMA tile t+1 into the other buffer (overlaps compute)
      const uint32_t boff = (uint32_t)((t + 1) & 1) * BUF_BYTES;
      const size_t   goff = (size_t)(t + 1) * 32;  // halves along K
      ASYNC_CP_B128(lA0 + boff, gA0 + goff);
      ASYNC_CP_B128(lA1 + boff, gA1 + goff);
      ASYNC_CP_B128(lB0 + boff, gB0 + goff);
      ASYNC_CP_B128(lB1 + boff, gB1 + goff);
    }

    // fragment loads per documented 16-bit A/B layout:
    // lane holds row (lane%16); elems 0..7 = K lh*8+0..7, elems 8..15 = K 16+lh*8+0..7
    const _Float16* Ab = &As[cur * TILE_HALVES];
    const _Float16* Bb = &Bs[cur * TILE_HALVES];
    v16h af[2], bf[4];
#pragma unroll
    for (int mi = 0; mi < 2; ++mi) {
      const _Float16* p = &Ab[(wm + mi * 16 + lr) * PAD + lh * 8];
      v8h lo = *(const v8h*)(p);
      v8h hi = *(const v8h*)(p + 16);
      af[mi] = __builtin_shufflevector(lo, hi, 0,1,2,3,4,5,6,7,8,9,10,11,12,13,14,15);
    }
#pragma unroll
    for (int ni = 0; ni < 4; ++ni) {
      const _Float16* p = &Bb[(wn + ni * 16 + lr) * PAD + lh * 8];
      v8h lo = *(const v8h*)(p);
      v8h hi = *(const v8h*)(p + 16);
      bf[ni] = __builtin_shufflevector(lo, hi, 0,1,2,3,4,5,6,7,8,9,10,11,12,13,14,15);
    }
#pragma unroll
    for (int mi = 0; mi < 2; ++mi)
#pragma unroll
      for (int ni = 0; ni < 4; ++ni)
        acc[mi][ni] = __builtin_amdgcn_wmma_f32_16x16x32_f16(
            false, af[mi], false, bf[ni], (short)0, acc[mi][ni], false, false);
  }

  // epilogue: D layout VGPR r -> M = r + 8*(lane/16); N = lane%16
#pragma unroll
  for (int mi = 0; mi < 2; ++mi) {
#pragma unroll
    for (int ni = 0; ni < 4; ++ni) {
#pragma unroll
      for (int r = 0; r < 8; ++r) {
        const int m = by + wm + mi * 16 + r + lh * 8;
        const int n = bx + wn + ni * 16 + lr;
        float v = acc[mi][ni][r] * scale;
        if (bias) v += bias[m];
        if constexpr (MODE == EP_F16N) {
          ((_Float16*)OutV)[(size_t)bz * sOut + (size_t)m * ldo + n] = (_Float16)v;
        } else if constexpr (MODE == EP_F16T) {
          ((_Float16*)OutV)[(size_t)bz * sOut + (size_t)n * ldo + m] = (_Float16)v;
        } else if constexpr (MODE == EP_F32N) {
          ((float*)OutV)[(size_t)bz * sOut + (size_t)m * ldo + n] = v;
        } else {  // EP_RESID
          ((float*)OutV)[(size_t)bz * sOut + (size_t)m * ldo + n] =
              v + res[(size_t)bz * sRes + (size_t)m * ldo + n];
        }
      }
    }
  }
}

// ---------------- softmax over a 1024-wide logits row, in-place f32 -> f16 ----------------
__global__ __launch_bounds__(256) void softmax_kernel(float* __restrict__ logits) {
  const size_t rowBase = ((size_t)blockIdx.y * N_PIX + blockIdx.x) * N_PIX;
  const float* src = logits + rowBase;
  _Float16* dst = (_Float16*)logits + rowBase * 2;  // first half of the row's byte span
  float v[4];
#pragma unroll
  for (int j = 0; j < 4; ++j) v[j] = src[threadIdx.x + j * 256];

  __shared__ float red[16];
  float mx = fmaxf(fmaxf(v[0], v[1]), fmaxf(v[2], v[3]));
  for (int off = 16; off > 0; off >>= 1) mx = fmaxf(mx, __shfl_down(mx, off, 32));
  const int wid = threadIdx.x >> 5, ln = threadIdx.x & 31;
  if (ln == 0) red[wid] = mx;
  __syncthreads();
  if (threadIdx.x == 0) {
    float m = red[0];
    for (int i = 1; i < 8; ++i) m = fmaxf(m, red[i]);
    red[0] = m;
  }
  __syncthreads();
  mx = red[0];

  float e[4], s = 0.f;
#pragma unroll
  for (int j = 0; j < 4; ++j) { e[j] = __expf(v[j] - mx); s += e[j]; }
  for (int off = 16; off > 0; off >>= 1) s += __shfl_down(s, off, 32);
  if (ln == 0) red[8 + wid] = s;
  __syncthreads();
  if (threadIdx.x == 0) {
    float t = 0.f;
    for (int i = 0; i < 8; ++i) t += red[8 + i];
    red[8] = t;
  }
  __syncthreads();
  const float inv = 1.f / red[8];
  __syncthreads();  // all row reads done (held in registers) before in-place overwrite
#pragma unroll
  for (int j = 0; j < 4; ++j) dst[threadIdx.x + j * 256] = (_Float16)(e[j] * inv);
}

// ---------------- launch ----------------
extern "C" void kernel_launch(void* const* d_in, const int* in_sizes, int n_in,
                              void* d_out, int out_size, void* d_ws, size_t ws_size,
                              hipStream_t stream) {
  const float* x     = (const float*)d_in[0];
  const float* gamma = (const float*)d_in[1];
  const float* beta  = (const float*)d_in[2];
  const float* wq    = (const float*)d_in[3];
  const float* bq    = (const float*)d_in[4];
  const float* wk    = (const float*)d_in[5];
  const float* bk    = (const float*)d_in[6];
  const float* wv    = (const float*)d_in[7];
  const float* bv    = (const float*)d_in[8];
  const float* wp    = (const float*)d_in[9];
  const float* bp    = (const float*)d_in[10];

  char* ws = (char*)d_ws;
  _Float16* h_t  = (_Float16*)(ws);                           // [B,N,C] f16, 16 MB
  _Float16* q_t  = (_Float16*)(ws + (size_t)16 * (1u << 20)); // [B,N,C] f16
  _Float16* k_t  = (_Float16*)(ws + (size_t)32 * (1u << 20)); // [B,N,C] f16
  _Float16* vbuf = (_Float16*)(ws + (size_t)48 * (1u << 20)); // [B,C,N] f16
  _Float16* o_t  = (_Float16*)(ws + (size_t)64 * (1u << 20)); // [B,N,C] f16
  _Float16* w16  = (_Float16*)(ws + (size_t)80 * (1u << 20)); // 4x [C,C] f16, 2 MB
  float*  logits = (float*)   (ws + (size_t)82 * (1u << 20)); // [B,N,N] f32 (attn f16 aliased), 64 MB

  const long long sBC = (long long)N_PIX * C_DIM;  // per-batch stride (elements)
  const float qscale  = 0.044194173824159216f;     // 512^-0.5 folded into q

  wcvt_kernel<<<dim3(1024), dim3(256), 0, stream>>>(wq, wk, wv, wp, w16);
  groupnorm_kernel<<<dim3(GROUPS, BATCH), dim3(256), 0, stream>>>(x, gamma, beta, h_t);

  // q_t[n,c] = (wq @ h) * scale + bq   (store transposed)
  gemm_nt_kernel<EP_F16T><<<dim3(N_PIX / 128, C_DIM / 128, BATCH), 256, 0, stream>>>(
      w16 + 0, 0, C_DIM, h_t, sBC, C_DIM, q_t, sBC, C_DIM, bq, nullptr, 0, C_DIM, qscale);
  // k_t[n,c] = wk @ h + bk
  gemm_nt_kernel<EP_F16T><<<dim3(N_PIX / 128, C_DIM / 128, BATCH), 256, 0, stream>>>(
      w16 + 262144, 0, C_DIM, h_t, sBC, C_DIM, k_t, sBC, C_DIM, bk, nullptr, 0, C_DIM, 1.f);
  // v[c,n] = wv @ h + bv   (channel-major)
  gemm_nt_kernel<EP_F16N><<<dim3(N_PIX / 128, C_DIM / 128, BATCH), 256, 0, stream>>>(
      w16 + 524288, 0, C_DIM, h_t, sBC, C_DIM, vbuf, sBC, N_PIX, bv, nullptr, 0, C_DIM, 1.f);
  // logits[n,m] = q_t[n,:] . k_t[m,:]
  gemm_nt_kernel<EP_F32N><<<dim3(N_PIX / 128, N_PIX / 128, BATCH), 256, 0, stream>>>(
      q_t, sBC, C_DIM, k_t, sBC, C_DIM, logits, (long long)N_PIX * N_PIX, N_PIX,
      nullptr, nullptr, 0, C_DIM, 1.f);
  // softmax rows, in-place f16
  softmax_kernel<<<dim3(N_PIX, BATCH), 256, 0, stream>>>(logits);
  // o_t[n,c] = attn[n,:] . v[c,:]   (attn f16 aliased over logits, row stride 2*N halves)
  gemm_nt_kernel<EP_F16N><<<dim3(C_DIM / 128, N_PIX / 128, BATCH), 256, 0, stream>>>(
      (const _Float16*)logits, (long long)N_PIX * N_PIX * 2, 2 * N_PIX,
      vbuf, sBC, N_PIX, o_t, sBC, C_DIM, nullptr, nullptr, 0, N_PIX, 1.f);
  // out[c,n] = x[c,n] + bp[c] + wp[c,:] . o_t[n,:]
  gemm_nt_kernel<EP_RESID><<<dim3(N_PIX / 128, C_DIM / 128, BATCH), 256, 0, stream>>>(
      w16 + 786432, 0, C_DIM, o_t, sBC, C_DIM, (float*)d_out, sBC, N_PIX,
      bp, x, sBC, C_DIM, 1.f);
}
